// GConv_88124138979802
// MI455X (gfx1250) — matrix operand order repaired and checked by reading, not compiled, not run on hardware
//
#include <hip/hip_runtime.h>

typedef __attribute__((ext_vector_type(16))) _Float16 v16h;
typedef __attribute__((ext_vector_type(8)))  float    v8f;

#define D_FEAT 128   // D_IN == D_HID == 128 in the reference

// ---- W (f32, row-major K x N) -> f16, pre-swizzled into WMMA B-fragment order ----
// Destination index i = ((nt*4 + kk32)*32 + lane)*16 + j maps to
// W[kk32*32 + 16*(j/8) + 8*(lane/16) + j%8][nt*16 + (lane%16)], so that at GEMM time a wave
// reading (nt, kk32) fetches per-lane 16 contiguous halves (= two b128 loads).
__global__ void __launch_bounds__(256) k_convert_w_swz(const float* __restrict__ W,
                                                       _Float16* __restrict__ Whs) {
  int i = blockIdx.x * 256 + threadIdx.x;       // 16384 elements
  int j    = i & 15;
  int lane = (i >> 4) & 31;
  int kk32 = (i >> 9) & 3;
  int nt   = i >> 11;
  int k = kk32 * 32 + 16 * (j >> 3) + 8 * (lane >> 4) + (j & 7);
  int n = nt * 16 + (lane & 15);
  Whs[i] = (_Float16)W[k * D_FEAT + n];
}

// ---------------- degree counting (int atomics, L2-resident) ----------------
__global__ void __launch_bounds__(256) k_degree(const int* __restrict__ src,
                                                const int* __restrict__ dst,
                                                int* __restrict__ cout_,
                                                int* __restrict__ cin_, int E) {
  int e = blockIdx.x * 256 + threadIdx.x;
  if (e < E) {
    atomicAdd(&cout_[src[e]], 1);
    atomicAdd(&cin_[dst[e]], 1);
  }
}

__global__ void __launch_bounds__(256) k_rsqrt(const int* __restrict__ c0, float* __restrict__ r0,
                                               const int* __restrict__ c1, float* __restrict__ r1,
                                               int N) {
  int i = blockIdx.x * 256 + threadIdx.x;
  if (i < N) {
    float a = (float)c0[i];
    float b = (float)c1[i];
    r0[i] = rsqrtf(a > 1.f ? a : 1.f);
    r1[i] = rsqrtf(b > 1.f ? b : 1.f);
  }
}

// ---------------- edge scatter: one wave32 per edge ----------------
// lane l reads float4 l of the 128-float source row (coalesced b128 loads),
// scales by deg_out^{-1/2}[src], and atomically adds into agg[dst].
// Working set (51.2MB src rows + 51.2MB agg) is L2-resident on MI455X (192MB L2).
__global__ void __launch_bounds__(256) k_scatter(const float* __restrict__ x,
                                                 const int* __restrict__ src,
                                                 const int* __restrict__ dst,
                                                 const float* __restrict__ dois,
                                                 float* __restrict__ agg, int E) {
  long long gid = (long long)blockIdx.x * 256 + threadIdx.x;
  int e    = (int)(gid >> 5);
  int lane = (int)(gid & 31);
  if (e >= E) return;
  int s = src[e];
  int d = dst[e];
  float sc = dois[s];
  float4 v = ((const float4*)(x + (long long)s * D_FEAT))[lane];
  float* ap = agg + (long long)d * D_FEAT + lane * 4;
  unsafeAtomicAdd(ap + 0, v.x * sc);   // global_atomic_add_f32
  unsafeAtomicAdd(ap + 1, v.y * sc);
  unsafeAtomicAdd(ap + 2, v.z * sc);
  unsafeAtomicAdd(ap + 3, v.w * sc);
}

// ---------------- GEMM + deg_in scaling + bias + ReLU, via WMMA ----------------
// One wave computes a full 16x128 output stripe: 8 C tiles (v8f c[8] = 64 VGPRs),
// A fragment loaded ONCE per 32-K step (8x less agg traffic than tile-per-wave),
// then 8 WMMAs against the swizzled, L0-resident B fragments.
// A frag (16x32 f16, ISA layout): lane L holds row M=L%16;
//   elems 0..7  = K 8*(L/16)+0..7,  elems 8..15 = K 16+8*(L/16)+0..7
// B frag (32x16 f16): lane holds col N=L%16; elem j = K 16*(j/8)+8*(L/16)+j%8  (pre-swizzled)
// C/D (16x16 f32): VGPR r, lane L -> M = r + 8*(L/16), N = L%16
__global__ void __launch_bounds__(256) k_gemm_bias_relu(const float* __restrict__ agg,
                                                        const float* __restrict__ diis,
                                                        const _Float16* __restrict__ Whs,
                                                        const float* __restrict__ bias,
                                                        float* __restrict__ out, int N) {
  int wid  = (blockIdx.x * 256 + threadIdx.x) >> 5;   // global wave id == m-tile
  int lane = threadIdx.x & 31;
  int m = wid * 16 + (lane & 15);
  if (m >= N) return;                  // wave-uniform (N % 16 == 0): EXEC stays all-1s
  int half = lane >> 4;
  int ncol = lane & 15;

  float s = diis[m];                   // fuse deg_in^{-1/2} row scaling into A load

  v8f c[8] = {};
  const float* arow = agg + (long long)m * D_FEAT;
  #pragma unroll
  for (int kk32 = 0; kk32 < 4; ++kk32) {
    v16h a;
    const float* ap = arow + kk32 * 32 + 8 * half;
    #pragma unroll
    for (int j = 0; j < 8; ++j) {
      a[j]     = (_Float16)(ap[j] * s);
      a[j + 8] = (_Float16)(ap[16 + j] * s);
    }
    #pragma unroll
    for (int nt = 0; nt < 8; ++nt) {
      v16h b = *(const v16h*)(Whs + (((nt * 4 + kk32) * 32 + lane) << 4));
      c[nt] = __builtin_amdgcn_wmma_f32_16x16x32_f16(false, a, false, b, (short)0,
                                                     c[nt], false, false);
    }
  }

  int rowbase = wid * 16 + 8 * half;
  #pragma unroll
  for (int nt = 0; nt < 8; ++nt) {
    float bn = bias[nt * 16 + ncol];
    #pragma unroll
    for (int r = 0; r < 8; ++r) {
      float v = c[nt][r] + bn;
      out[(long long)(rowbase + r) * D_FEAT + nt * 16 + ncol] = v > 0.f ? v : 0.f;
    }
  }
}

extern "C" void kernel_launch(void* const* d_in, const int* in_sizes, int n_in,
                              void* d_out, int out_size, void* d_ws, size_t ws_size,
                              hipStream_t stream) {
  const float* feat = (const float*)d_in[0];
  const int*   src  = (const int*)d_in[1];
  const int*   dst  = (const int*)d_in[2];
  const float* W1   = (const float*)d_in[3];
  const float* b1   = (const float*)d_in[4];
  const float* W2   = (const float*)d_in[5];
  const float* b2   = (const float*)d_in[6];
  float* out = (float*)d_out;

  const int N = in_sizes[0] / D_FEAT;   // 100000
  const int E = in_sizes[1];            // 1600000

  // ---- workspace carve-up (~53 MB) ----
  char* ws = (char*)d_ws;
  size_t off = 0;
  auto carve = [&](size_t bytes) -> void* {
    void* p = ws + off;
    off += (bytes + 255) & ~(size_t)255;
    return p;
  };
  float*    agg  = (float*)carve((size_t)N * D_FEAT * sizeof(float));
  int*      co   = (int*)carve((size_t)N * sizeof(int));
  int*      ci   = (int*)carve((size_t)N * sizeof(int));
  float*    dois = (float*)carve((size_t)N * sizeof(float));
  float*    diis = (float*)carve((size_t)N * sizeof(float));
  _Float16* W1h  = (_Float16*)carve((size_t)D_FEAT * D_FEAT * sizeof(_Float16));
  _Float16* W2h  = (_Float16*)carve((size_t)D_FEAT * D_FEAT * sizeof(_Float16));

  const int nW      = D_FEAT * D_FEAT;
  const int blkW    = (nW + 255) / 256;
  const int blkE    = (E + 255) / 256;
  const int blkN    = (N + 255) / 256;
  const int blkScat = (int)(((long long)E * 32 + 255) / 256);
  const int nWaves  = (N + 15) / 16;                 // one wave per 16-row stripe
  const int blkGemm = (nWaves * 32 + 255) / 256;     // 8 waves per block

  // weights -> f16 in WMMA B-fragment order
  k_convert_w_swz<<<blkW, 256, 0, stream>>>(W1, W1h);
  k_convert_w_swz<<<blkW, 256, 0, stream>>>(W2, W2h);

  // degrees -> rsqrt norms
  hipMemsetAsync(co, 0, (size_t)N * sizeof(int), stream);
  hipMemsetAsync(ci, 0, (size_t)N * sizeof(int), stream);
  k_degree<<<blkE, 256, 0, stream>>>(src, dst, co, ci, E);
  k_rsqrt<<<blkN, 256, 0, stream>>>(co, dois, ci, diis, N);

  // ---- layer 1: agg = scatter(feat * dois); out = relu((agg*diis) @ W1 + b1) ----
  hipMemsetAsync(agg, 0, (size_t)N * D_FEAT * sizeof(float), stream);
  k_scatter<<<blkScat, 256, 0, stream>>>(feat, src, dst, dois, agg, E);
  k_gemm_bias_relu<<<blkGemm, 256, 0, stream>>>(agg, diis, W1h, b1, out, N);

  // ---- layer 2: agg = scatter(out * dois); out = relu((agg*diis) @ W2 + b2) ----
  hipMemsetAsync(agg, 0, (size_t)N * D_FEAT * sizeof(float), stream);
  k_scatter<<<blkScat, 256, 0, stream>>>(out, src, dst, dois, agg, E);
  k_gemm_bias_relu<<<blkGemm, 256, 0, stream>>>(agg, diis, W2h, b2, out, N);
}
